// TemporalAttention_21930103013487
// MI455X (gfx1250) — compile-verified, matrix operand order checked by reading, not run on hardware
//
#include <hip/hip_runtime.h>
#include <hip/hip_bf16.h>
#include <math.h>

// ---------------- problem constants ----------------
constexpr int kS = 32, kB = 128, kT = 128, kH = 1024;
constexpr int kNIN = 128, kNEXT = 8, kOUT = 128;
constexpr int kMB = kS * kB;              // 4096 fused (s,b) rows
constexpr int kKCAT = kNIN + kNEXT + kH;  // 1160
constexpr int kKPAD = 1184;               // padded to 37 * 32

// ---------------- WMMA types (gfx1250, wave32) ----------------
typedef __attribute__((ext_vector_type(16))) __bf16        v16bf;
typedef __attribute__((ext_vector_type(8)))  __bf16        v8bf;
typedef __attribute__((ext_vector_type(8)))  float         v8f;

__device__ __forceinline__ unsigned pk2(float a, float b) {
  // fp32 -> bf16 RNE, pack two into one dword (used only in prep kernels)
  unsigned ua = __builtin_bit_cast(unsigned, a);
  unsigned ub = __builtin_bit_cast(unsigned, b);
  ua = (ua + 0x7FFFu + ((ua >> 16) & 1u)) >> 16;
  ub = (ub + 0x7FFFu + ((ub >> 16) & 1u)) >> 16;
  return ua | (ub << 16);
}

__device__ __forceinline__ unsigned short cvt1(float a) {
  unsigned u = __builtin_bit_cast(unsigned, a);
  u = (u + 0x7FFFu + ((u >> 16) & 1u)) >> 16;
  return (unsigned short)u;
}

// gfx1250 has V_TANH_F32 (TRANS op, co-executes with VALU)
__device__ __forceinline__ float fast_tanh(float x) {
#if __has_builtin(__builtin_amdgcn_tanhf)
  return __builtin_amdgcn_tanhf(x);
#else
  return tanhf(x);
#endif
}

__device__ __forceinline__ v8f wmma_bf16(v16bf a, v16bf b, v8f c) {
  return __builtin_amdgcn_wmma_f32_16x16x32_bf16(false, a, false, b,
                                                 (short)0, c, false, false);
}

// bf16 fragment: K contiguous along the lane's row. Two 16B loads, no VALU cvt.
// VGPR j holds K = (j>=4?16:0) + (lane>=16?8:0) + 2*(j&3) + {0,1}
__device__ __forceinline__ v16bf frag_bf(const unsigned short* __restrict__ rowbase,
                                         int k0, int hh) {
  const unsigned short* q = rowbase + k0 + hh * 8;
  v8bf lo = *reinterpret_cast<const v8bf*>(q);       // K = k0+hh*8 .. +7
  v8bf hi = *reinterpret_cast<const v8bf*>(q + 16);  // K = k0+16+hh*8 .. +7
  return __builtin_shufflevector(lo, hi, 0, 1, 2, 3, 4, 5, 6, 7,
                                 8, 9, 10, 11, 12, 13, 14, 15);
}

__device__ __forceinline__ float sigm(float x) { return 1.f / (1.f + __expf(-x)); }

// ================= prep kernels (bandwidth-bound, one-time conversions) =================

__global__ void __launch_bounds__(256)
k_cvt(const float* __restrict__ in, unsigned* __restrict__ out, int npairs) {
  int i = blockIdx.x * 256 + threadIdx.x;
  if (i < npairs) out[i] = pk2(in[2 * i], in[2 * i + 1]);
}

// hiddenT[b][t][c] (bf16, c contiguous) = AS[b][c][t]; LDS-tiled 32x32 transpose
__global__ void __launch_bounds__(256)
k_transpose(const float* __restrict__ AS, unsigned* __restrict__ hidT) {
  __shared__ float tile[32][33];
  const int b = blockIdx.z, c0 = blockIdx.x * 32, t0 = blockIdx.y * 32;
  const int tt = threadIdx.x & 31, ci = threadIdx.x >> 5;
  const float* src = AS + ((size_t)b * kH + c0) * kT + t0;
#pragma unroll
  for (int it = 0; it < 4; ++it) {
    int cc = ci + it * 8;
    tile[cc][tt] = src[(size_t)cc * kT + tt];  // coalesced over t
  }
  __syncthreads();
  const int cp = threadIdx.x & 15;
  const int ti = threadIdx.x >> 4;
  unsigned* dst = hidT + ((size_t)b * kT + t0) * (kH / 2) + c0 / 2;
#pragma unroll
  for (int it = 0; it < 2; ++it) {
    int t = ti + it * 16;
    dst[(size_t)t * (kH / 2) + cp] = pk2(tile[2 * cp][t], tile[2 * cp + 1][t]);
  }
}

__device__ __forceinline__ float geta_cat(const float* dec, const float* ext,
                                          const float* ai, int m, int k) {
  if (k < kNIN) return dec[(size_t)m * kNIN + k];
  if (k < kNIN + kNEXT) return ext[(size_t)m * kNEXT + (k - kNIN)];
  if (k < kKCAT) return ai[(size_t)m * kH + (k - kNIN - kNEXT)];
  return 0.f;
}

__global__ void __launch_bounds__(256)
k_catpad(const float* __restrict__ dec, const float* __restrict__ ext,
         const float* __restrict__ ainit, unsigned* __restrict__ xcat) {
  int idx = blockIdx.x * 256 + threadIdx.x;
  if (idx >= kMB * (kKPAD / 2)) return;
  int m = idx / (kKPAD / 2), k = (idx % (kKPAD / 2)) * 2;
  xcat[idx] = pk2(geta_cat(dec, ext, ainit, m, k), geta_cat(dec, ext, ainit, m, k + 1));
}

__global__ void __launch_bounds__(256)
k_padw(const float* __restrict__ Wcc, unsigned* __restrict__ out) {
  int idx = blockIdx.x * 256 + threadIdx.x;
  if (idx >= kNIN * (kKPAD / 2)) return;
  int n = idx / (kKPAD / 2), k = (idx % (kKPAD / 2)) * 2;
  float a0 = (k < kKCAT) ? Wcc[(size_t)n * kKCAT + k] : 0.f;
  float a1 = (k + 1 < kKCAT) ? Wcc[(size_t)n * kKCAT + k + 1] : 0.f;
  out[idx] = pk2(a0, a1);
}

// ============ K1: hf[b,o,t] = sum_c Wconv[o,c]*hidden[b,c,t] + bconv[o] ============
// A = Wc_bf rows(o); B[k=c][n=t] = hiddenT[b][t][c] rows(t). Wave: 64(o) x 32(t).
__global__ void __launch_bounds__(256)
k_conv(const unsigned short* __restrict__ WcB, const unsigned short* __restrict__ hidT,
       const float* __restrict__ bc, float* __restrict__ hf) {
  const int lane = threadIdx.x & 31, w = threadIdx.x >> 5;
  const int hh = lane >> 4, r = lane & 15;
  const int b = blockIdx.z;
  const int t0 = blockIdx.y * 32;
  const int o_base = blockIdx.x * 512 + w * 64;
  const unsigned short* hT = hidT + ((size_t)b * kT + t0) * kH;

  // loop-invariant row bases, hoisted so the K-loop has no address math
  const unsigned short* bp0 = hT + (size_t)r * kH;
  const unsigned short* bp1 = hT + (size_t)(16 + r) * kH;
  const unsigned short* ap0 = WcB + (size_t)(o_base + r) * kH;
  const unsigned short* ap1 = ap0 + (size_t)16 * kH;
  const unsigned short* ap2 = ap0 + (size_t)32 * kH;
  const unsigned short* ap3 = ap0 + (size_t)48 * kH;

  v8f z = {};
  v8f acc[8] = {z, z, z, z, z, z, z, z};
  for (int k0 = 0; k0 < kH; k0 += 32) {
    v16bf bf0 = frag_bf(bp0, k0, hh);
    v16bf bf1 = frag_bf(bp1, k0, hh);
    v16bf a0 = frag_bf(ap0, k0, hh);
    v16bf a1 = frag_bf(ap1, k0, hh);
    v16bf a2 = frag_bf(ap2, k0, hh);
    v16bf a3 = frag_bf(ap3, k0, hh);
    acc[0] = wmma_bf16(a0, bf0, acc[0]);
    acc[1] = wmma_bf16(a0, bf1, acc[1]);
    acc[2] = wmma_bf16(a1, bf0, acc[2]);
    acc[3] = wmma_bf16(a1, bf1, acc[3]);
    acc[4] = wmma_bf16(a2, bf0, acc[4]);
    acc[5] = wmma_bf16(a2, bf1, acc[5]);
    acc[6] = wmma_bf16(a3, bf0, acc[6]);
    acc[7] = wmma_bf16(a3, bf1, acc[7]);
  }
#pragma unroll
  for (int i = 0; i < 4; ++i)
#pragma unroll
    for (int j = 0; j < 2; ++j)
#pragma unroll
      for (int rr = 0; rr < 8; ++rr) {
        int o = o_base + i * 16 + rr + 8 * hh;
        hf[(size_t)b * kH * kT + (size_t)o * kT + t0 + j * 16 + r] =
            acc[i * 2 + j][rr] + bc[o];
      }
}

// ============ K2a: x = xcat @ Wcat^T + bcat (bf16 out). Wave: 32m x 16n ============
__global__ void __launch_bounds__(256)
k_concat_x(const unsigned short* __restrict__ xcat, const unsigned short* __restrict__ Wcc,
           const float* __restrict__ bcc, unsigned short* __restrict__ x) {
  const int lane = threadIdx.x & 31, w = threadIdx.x >> 5;
  const int hh = lane >> 4, r = lane & 15;
  const int m0 = (blockIdx.x * 8 + w) * 32;
  const int n0 = blockIdx.y * 16;
  const unsigned short* a0p = xcat + (size_t)(m0 + r) * kKPAD;
  const unsigned short* a1p = a0p + (size_t)16 * kKPAD;
  const unsigned short* bp  = Wcc + (size_t)(n0 + r) * kKPAD;
  v8f a0 = {}, a1 = {};
  for (int k0 = 0; k0 < kKPAD; k0 += 32) {
    v16bf f0 = frag_bf(a0p, k0, hh);
    v16bf f1 = frag_bf(a1p, k0, hh);
    v16bf bf = frag_bf(bp, k0, hh);
    a0 = wmma_bf16(f0, bf, a0);
    a1 = wmma_bf16(f1, bf, a1);
  }
  const float bias = bcc[n0 + r];
#pragma unroll
  for (int rr = 0; rr < 8; ++rr) {
    x[(size_t)(m0 + rr + 8 * hh) * kNIN + n0 + r]      = cvt1(a0[rr] + bias);
    x[(size_t)(m0 + 16 + rr + 8 * hh) * kNIN + n0 + r] = cvt1(a1[rr] + bias);
  }
}

// ============ K2b: i,g,o gates -> c, h (bf16), state (f32). Wave: 32m x 16n ============
__global__ void __launch_bounds__(256)
k_lstm(const unsigned short* __restrict__ x, const unsigned short* __restrict__ Wih,
       const float* __restrict__ bih, const float* __restrict__ bhh,
       unsigned short* __restrict__ c, unsigned short* __restrict__ h,
       float* __restrict__ state_out) {
  const int lane = threadIdx.x & 31, w = threadIdx.x >> 5;
  const int hh = lane >> 4, r = lane & 15;
  const int m0 = (blockIdx.x * 8 + w) * 32;
  const int j0 = blockIdx.y * 16;
  const unsigned short* a0p = x + (size_t)(m0 + r) * kNIN;
  const unsigned short* a1p = a0p + (size_t)16 * kNIN;
  const unsigned short* bip = Wih + (size_t)(j0 + r) * kNIN;
  const unsigned short* bgp = bip + (size_t)2048 * kNIN;
  const unsigned short* bop = bip + (size_t)3072 * kNIN;
  v8f z = {};
  v8f ai[2] = {z, z}, ag[2] = {z, z}, ao[2] = {z, z};
  for (int k0 = 0; k0 < kNIN; k0 += 32) {
    v16bf bi = frag_bf(bip, k0, hh);
    v16bf bg = frag_bf(bgp, k0, hh);
    v16bf bo = frag_bf(bop, k0, hh);
    v16bf f0 = frag_bf(a0p, k0, hh);
    v16bf f1 = frag_bf(a1p, k0, hh);
    ai[0] = wmma_bf16(f0, bi, ai[0]);
    ag[0] = wmma_bf16(f0, bg, ag[0]);
    ao[0] = wmma_bf16(f0, bo, ao[0]);
    ai[1] = wmma_bf16(f1, bi, ai[1]);
    ag[1] = wmma_bf16(f1, bg, ag[1]);
    ao[1] = wmma_bf16(f1, bo, ao[1]);
  }
  const int n = j0 + r;
  const float bi_ = bih[n] + bhh[n];
  const float bg_ = bih[2048 + n] + bhh[2048 + n];
  const float bo_ = bih[3072 + n] + bhh[3072 + n];
#pragma unroll
  for (int im = 0; im < 2; ++im)
#pragma unroll
    for (int rr = 0; rr < 8; ++rr) {
      int m = m0 + im * 16 + rr + 8 * hh;
      float cc = sigm(ai[im][rr] + bi_) * fast_tanh(ag[im][rr] + bg_);
      float hv = sigm(ao[im][rr] + bo_) * fast_tanh(cc);
      c[(size_t)m * kH + n] = cvt1(cc);
      h[(size_t)m * kH + n] = cvt1(hv);
      if (m >= (kS - 1) * kB)  // s == 31 -> final cell state (full f32 from accumulator)
        state_out[(size_t)(m - (kS - 1) * kB) * kH + n] = cc;
    }
}

// ============ K2c: y = c @ W_attn^T + b_attn (f32 out). Wave: 32m x 32n ============
__global__ void __launch_bounds__(256)
k_query(const unsigned short* __restrict__ c, const unsigned short* __restrict__ Wa,
        const float* __restrict__ ba, float* __restrict__ y) {
  const int lane = threadIdx.x & 31, w = threadIdx.x >> 5;
  const int hh = lane >> 4, r = lane & 15;
  const int m0 = (blockIdx.x * 8 + w) * 32;
  const int n0 = blockIdx.y * 32;
  const unsigned short* a0p = c + (size_t)(m0 + r) * kH;
  const unsigned short* a1p = a0p + (size_t)16 * kH;
  const unsigned short* b0p = Wa + (size_t)(n0 + r) * kH;
  const unsigned short* b1p = b0p + (size_t)16 * kH;
  v8f z = {};
  v8f acc[4] = {z, z, z, z};
  for (int k0 = 0; k0 < kH; k0 += 32) {
    v16bf a0 = frag_bf(a0p, k0, hh);
    v16bf a1 = frag_bf(a1p, k0, hh);
    v16bf b0 = frag_bf(b0p, k0, hh);
    v16bf b1 = frag_bf(b1p, k0, hh);
    acc[0] = wmma_bf16(a0, b0, acc[0]);
    acc[1] = wmma_bf16(a0, b1, acc[1]);
    acc[2] = wmma_bf16(a1, b0, acc[2]);
    acc[3] = wmma_bf16(a1, b1, acc[3]);
  }
#pragma unroll
  for (int im = 0; im < 2; ++im)
#pragma unroll
    for (int in = 0; in < 2; ++in)
#pragma unroll
      for (int rr = 0; rr < 8; ++rr)
        y[(size_t)(m0 + im * 16 + rr + 8 * hh) * kH + n0 + in * 16 + r] =
            acc[im * 2 + in][rr] + ba[n0 + in * 16 + r];
}

// ============ K3: logits -> softmax -> context (TRANS/VALU-bound) ============
__global__ void __launch_bounds__(256)
k_attn(const float* __restrict__ hf, const float* __restrict__ AS,
       const float* __restrict__ vvec, const float* __restrict__ y,
       unsigned short* __restrict__ dall) {
  const int sb = blockIdx.x;  // fused (s,b)
  const int b = sb & (kB - 1);
  __shared__ float part[256];
  __shared__ float aw[kT];
  __shared__ float red[2];
  const int tid = threadIdx.x;
  const int t = tid & (kT - 1), half = tid >> 7;
  const float* hfb = hf + (size_t)b * kH * kT;
  const float* yb = y + (size_t)sb * kH;

  float accp = 0.f;
  const int cbeg = half * (kH / 2);
  for (int cc = cbeg; cc < cbeg + kH / 2; ++cc)
    accp += vvec[cc] * fast_tanh(hfb[(size_t)cc * kT + t] + yb[cc]);
  part[tid] = accp;
  __syncthreads();
  if (tid < kT) aw[t] = part[t] + part[t + kT];
  __syncthreads();
  if (tid == 0) {
    float mx = aw[0];
    for (int i = 1; i < kT; ++i) mx = fmaxf(mx, aw[i]);
    red[0] = mx;
  }
  __syncthreads();
  if (tid < kT) aw[t] = __expf(aw[t] - red[0]);
  __syncthreads();
  if (tid == 0) {
    float sum = 0.f;
    for (int i = 0; i < kT; ++i) sum += aw[i];
    red[1] = 1.f / sum;
  }
  __syncthreads();
  if (tid < kT) aw[t] *= red[1];
  __syncthreads();

  const float* hidb = AS + (size_t)b * kH * kT;
  unsigned short* dout = dall + (size_t)sb * kH;
#pragma unroll
  for (int j = 0; j < 4; ++j) {
    const int cc = tid + j * 256;
    const float* row = hidb + (size_t)cc * kT;
    float acc = 0.f;
    for (int tt = 0; tt < kT; tt += 4) {
      float4 hv = *reinterpret_cast<const float4*>(row + tt);
      acc += aw[tt] * hv.x + aw[tt + 1] * hv.y + aw[tt + 2] * hv.z + aw[tt + 3] * hv.w;
    }
    dout[cc] = cvt1(acc);
  }
}

// ============ K4: out = [h | d_all] @ W_out^T + b_out. Wave: 32m x 16n ============
__global__ void __launch_bounds__(256)
k_out(const unsigned short* __restrict__ h, const unsigned short* __restrict__ dall,
      const unsigned short* __restrict__ Wo, const float* __restrict__ bo,
      float* __restrict__ out) {
  const int lane = threadIdx.x & 31, w = threadIdx.x >> 5;
  const int hh = lane >> 4, r = lane & 15;
  const int m0 = (blockIdx.x * 8 + w) * 32;
  const int n0 = blockIdx.y * 16;
  const unsigned short* h0p = h + (size_t)(m0 + r) * kH;
  const unsigned short* h1p = h0p + (size_t)16 * kH;
  const unsigned short* d0p = dall + (size_t)(m0 + r) * kH;
  const unsigned short* d1p = d0p + (size_t)16 * kH;
  const unsigned short* bp  = Wo + (size_t)(n0 + r) * (2 * kH);
  v8f a0 = {}, a1 = {};
  for (int k0 = 0; k0 < kH; k0 += 32) {  // h half
    v16bf bf = frag_bf(bp, k0, hh);
    a0 = wmma_bf16(frag_bf(h0p, k0, hh), bf, a0);
    a1 = wmma_bf16(frag_bf(h1p, k0, hh), bf, a1);
  }
  for (int k0 = 0; k0 < kH; k0 += 32) {  // d half
    v16bf bf = frag_bf(bp + kH, k0, hh);
    a0 = wmma_bf16(frag_bf(d0p, k0, hh), bf, a0);
    a1 = wmma_bf16(frag_bf(d1p, k0, hh), bf, a1);
  }
  const float bias = bo[n0 + r];
#pragma unroll
  for (int rr = 0; rr < 8; ++rr) {
    out[(size_t)(m0 + rr + 8 * hh) * kOUT + n0 + r]      = a0[rr] + bias;
    out[(size_t)(m0 + 16 + rr + 8 * hh) * kOUT + n0 + r] = a1[rr] + bias;
  }
}

// ---------------- host launcher ----------------
extern "C" void kernel_launch(void* const* d_in, const int* in_sizes, int n_in,
                              void* d_out, int out_size, void* d_ws, size_t ws_size,
                              hipStream_t stream) {
  (void)in_sizes; (void)n_in; (void)out_size; (void)ws_size;
  const float* dec   = (const float*)d_in[0];
  const float* ext   = (const float*)d_in[1];
  /* d_in[2] encoder_state: unused by the faithful forward */
  const float* AS    = (const float*)d_in[3];
  const float* ainit = (const float*)d_in[4];
  const float* Wc    = (const float*)d_in[5];
  const float* bc    = (const float*)d_in[6];
  const float* vvec  = (const float*)d_in[7];
  const float* Wa    = (const float*)d_in[8];
  const float* ba    = (const float*)d_in[9];
  const float* Wcc   = (const float*)d_in[10];
  const float* bcc   = (const float*)d_in[11];
  const float* Wih   = (const float*)d_in[12];
  const float* bih   = (const float*)d_in[13];
  const float* bhh   = (const float*)d_in[14];
  const float* Wo    = (const float*)d_in[15];
  const float* bo    = (const float*)d_in[16];

  float* out       = (float*)d_out;             // [S,B,OUT]
  float* state_out = out + (size_t)kMB * kOUT;  // [B,H]

  // workspace carve-up (4-byte units; all chunks keep 16B alignment)
  unsigned* ws = (unsigned*)d_ws;
  float*    hf      = (float*)ws;                         ws += (size_t)kB * kH * kT;       // f32
  float*    y       = (float*)ws;                         ws += (size_t)kMB * kH;           // f32
  unsigned* hidT_u  = ws;                                 ws += (size_t)kB * kT * kH / 2;   // bf16
  unsigned* WcB_u   = ws;                                 ws += (size_t)kH * kH / 2;
  unsigned* xcat_u  = ws;                                 ws += (size_t)kMB * kKPAD / 2;
  unsigned* WccB_u  = ws;                                 ws += (size_t)kNIN * kKPAD / 2;
  unsigned* xB_u    = ws;                                 ws += (size_t)kMB * kNIN / 2;
  unsigned* WihB_u  = ws;                                 ws += (size_t)(4 * kH) * kNIN / 2;
  unsigned* cB_u    = ws;                                 ws += (size_t)kMB * kH / 2;
  unsigned* hB_u    = ws;                                 ws += (size_t)kMB * kH / 2;
  unsigned* WaB_u   = ws;                                 ws += (size_t)kH * kH / 2;
  unsigned* dB_u    = ws;                                 ws += (size_t)kMB * kH / 2;
  unsigned* WoB_u   = ws;                                 ws += (size_t)kOUT * 2 * kH / 2;

  dim3 blk(256);
  // ---- one-time bf16 conversions (bandwidth-bound) ----
  k_cvt<<<dim3(2048), blk, 0, stream>>>(Wc, WcB_u, kH * kH / 2);
  k_cvt<<<dim3(1024), blk, 0, stream>>>(Wih, WihB_u, 4 * kH * kNIN / 2);
  k_cvt<<<dim3(2048), blk, 0, stream>>>(Wa, WaB_u, kH * kH / 2);
  k_cvt<<<dim3(512),  blk, 0, stream>>>(Wo, WoB_u, kOUT * 2 * kH / 2);
  k_transpose<<<dim3(kH / 32, kT / 32, kB), blk, 0, stream>>>(AS, hidT_u);
  k_catpad<<<dim3((kMB * kKPAD / 2 + 255) / 256), blk, 0, stream>>>(dec, ext, ainit, xcat_u);
  k_padw<<<dim3((kNIN * kKPAD / 2 + 255) / 256), blk, 0, stream>>>(Wcc, WccB_u);

  // ---- WMMA GEMM pipeline ----
  k_conv<<<dim3(2, 4, kB), blk, 0, stream>>>((const unsigned short*)WcB_u,
                                             (const unsigned short*)hidT_u, bc, hf);
  k_concat_x<<<dim3(16, 8), blk, 0, stream>>>((const unsigned short*)xcat_u,
                                              (const unsigned short*)WccB_u, bcc,
                                              (unsigned short*)xB_u);
  k_lstm<<<dim3(16, 64), blk, 0, stream>>>((const unsigned short*)xB_u,
                                           (const unsigned short*)WihB_u, bih, bhh,
                                           (unsigned short*)cB_u, (unsigned short*)hB_u,
                                           state_out);
  k_query<<<dim3(16, 32), blk, 0, stream>>>((const unsigned short*)cB_u,
                                            (const unsigned short*)WaB_u, ba, y);
  k_attn<<<dim3(kMB), blk, 0, stream>>>(hf, AS, vvec, y, (unsigned short*)dB_u);
  k_out<<<dim3(16, 8), blk, 0, stream>>>((const unsigned short*)hB_u,
                                         (const unsigned short*)dB_u,
                                         (const unsigned short*)WoB_u, bo, out);
}
